// Model_86474871538503
// MI455X (gfx1250) — compile-verified, matrix operand order checked by reading, not run on hardware
//
#include <hip/hip_runtime.h>
#include <hip/hip_bf16.h>

typedef __attribute__((ext_vector_type(16))) __bf16 v16bf;
typedef __attribute__((ext_vector_type(8)))  __bf16 v8bf;
typedef __attribute__((ext_vector_type(8)))  float  v8f;

#define NEG_SLOPE (11.0f/48.0f)
#define TAU_F 0.1f

// Problem constants
#define BB   16
#define TT   4096
#define CC   256
#define RR   8
#define WW   128
#define T1L  4110   // layer-1 input length (4096+15 pad, minus 1 from layer 0)
#define T2L  4108   // layer-2 input length
#define T3L  4104   // pre length

// ---------------------------------------------------------------------------
// Repack conv weights (Co, 256, 2) f32 -> tap-concatenated (Co, 512) bf16
// k < 256 -> tap 0 (time t), k >= 256 -> tap 1 (time t + dil)
// ---------------------------------------------------------------------------
__global__ void k_prep_wcat(const float* __restrict__ w, __bf16* __restrict__ wcat) {
    int idx = blockIdx.x * 256 + threadIdx.x;   // 256*512 total
    int co  = idx >> 9;
    int k   = idx & 511;
    int ci  = k & 255;
    int tap = k >> 8;
    wcat[idx] = (__bf16)w[(co * 256 + ci) * 2 + tap];
}

// ---------------------------------------------------------------------------
// Layer 0: Cin=1, K=2, dil=1, left-pad 15.  Output in (b, t, c) bf16 layout.
// ---------------------------------------------------------------------------
__global__ void k_layer0(const float* __restrict__ x, const float* __restrict__ w0,
                         const float* __restrict__ b0, __bf16* __restrict__ out) {
    int t = blockIdx.x;          // [0, T1L)
    int b = blockIdx.y;
    int c = threadIdx.x;         // [0, 256)
    float h0a = (t     >= 15) ? x[b * TT + t     - 15] : 0.f;
    float h0b = (t + 1 >= 15) ? x[b * TT + t + 1 - 15] : 0.f;
    float y = w0[c * 2 + 0] * h0a + w0[c * 2 + 1] * h0b + b0[c];
    y = (y >= 0.f) ? y : NEG_SLOPE * y;
    out[((size_t)b * T1L + t) * CC + c] = (__bf16)y;
}

// ---------------------------------------------------------------------------
// WMMA conv: y(256 x Tout) = Wcat(256 x 512) * Xcat(512 x Tout) per batch,
// Xcat[k][t] = in[b][t + (k>=256 ? dil : 0)][k & 255]   (in is (b, Tin, 256))
//
// Tiling: waves split along N(t): each wave owns one unique 16-t strip and
// iterates 4 M-tiles (64 co, selected by blockIdx.y).  The big activation
// tensor is loaded exactly once per wave (1 B-load per 4 WMMAs, no cross-wave
// duplication); the 256KB weight matrix is the redundantly-read operand and
// lives in L2.  bf16 x bf16 -> f32 accumulate via v_wmma_f32_16x16x32_bf16.
// ---------------------------------------------------------------------------
__global__ void __launch_bounds__(256)
k_conv_wmma(const __bf16* __restrict__ in, const __bf16* __restrict__ wcat,
            const float* __restrict__ bias,
            __bf16* __restrict__ out_bf, float* __restrict__ out_f32,
            int Tin, int Tout, int dil) {
    const int lane = threadIdx.x & 31;
    const int wave = threadIdx.x >> 5;
    const int b    = blockIdx.z;
    const int co0  = blockIdx.y * 64;    // 4 co quarters
    const int hlf  = lane >> 4;          // lane half selects 16-wide K sub-block
    const int l16  = lane & 15;

    int t = blockIdx.x * 128 + wave * 16 + l16;
    const bool tvalid = (t < Tout);
    if (t >= Tout) t = Tout - 1;         // uniform clamp: EXEC stays all-ones

    const __bf16* inb = in + (size_t)b * Tin * CC;
    const __bf16* bp0 = inb + (size_t)t * CC;       // tap 0 row (time t)
    const __bf16* bp1 = bp0 + (size_t)dil * CC;     // tap 1 row (time t+dil)

    v8f acc[4] = {};

    #pragma unroll 4
    for (int kk = 0; kk < 16; ++kk) {               // K = 512, 32 per WMMA
        const int k0    = kk * 32;
        const int cbase = (k0 & 255) + hlf * 16;
        const __bf16* bp = (k0 >= 256) ? bp1 : bp0;

        // B fragment: 32 k x 16 t; lane = (t column, K half), contiguous chans
        v16bf bfrag = *(const v16bf*)(bp + cbase);

        if (kk + 1 < 16) {                          // cover latency of next tap row
            const int k0n = (kk + 1) * 32;
            const __bf16* bpn = (k0n >= 256) ? bp1 : bp0;
            __builtin_prefetch(bpn + ((k0n & 255) + hlf * 16), 0, 1);
        }

        #pragma unroll
        for (int m = 0; m < 4; ++m) {
            // A fragment: 16 co rows x 32 k (row co0+m*16+l16, contiguous 16 bf16)
            v16bf afrag = *(const v16bf*)(wcat +
                (size_t)(co0 + m * 16 + l16) * 512 + k0 + hlf * 16);
            acc[m] = __builtin_amdgcn_wmma_f32_16x16x32_bf16(
                         false, afrag, false, bfrag, (short)0, acc[m], false, false);
        }
    }

    if (!tvalid) return;

    // Epilogue: D layout => VGPR v, lane half h: co = co0+m*16 + h*8 + v.
    // 8 consecutive co per lane -> one packed 16B bf16 store / 32B f32 store.
    #pragma unroll
    for (int m = 0; m < 4; ++m) {
        const int cob = co0 + m * 16 + hlf * 8;
        v8bf ob;
        v8f  of;
        #pragma unroll
        for (int v = 0; v < 8; ++v) {
            float y = acc[m][v] + bias[cob + v];
            y = (y >= 0.f) ? y : NEG_SLOPE * y;
            of[v] = y;
            ob[v] = (__bf16)y;
        }
        size_t base = ((size_t)b * Tout + t) * CC + cob;
        if (out_bf)  *(v8bf*)(out_bf  + base) = ob;
        if (out_f32) *(v8f*)(out_f32 + base) = of;
    }
}

// ---------------------------------------------------------------------------
// Head: layer-3 conv (R=8) + h^2 normalize + gumbel softmax + mus einsums.
// One wave per (b, t); lanes split the 256 channels; shfl_xor reductions.
// Writes gen_z and mus into d_out.
// ---------------------------------------------------------------------------
__global__ void __launch_bounds__(256)
k_head(const float* __restrict__ pre,          // (B, T3L, 256) f32
       const float* __restrict__ w3, const float* __restrict__ b3,
       const float* __restrict__ pcw, const float* __restrict__ pcb,
       const float* __restrict__ gumbel,       // (B, T, R)
       float* __restrict__ out_mus,            // (B, T)
       float* __restrict__ out_z) {            // (B, R, T)
    __shared__ float sw3[RR * CC * 2];
    __shared__ float spcw[RR * CC];
    __shared__ float sb3[RR], spcb[RR];

    const int tid = threadIdx.x;
    for (int i = tid; i < RR * CC * 2; i += 256) sw3[i]  = w3[i];
    for (int i = tid; i < RR * CC;     i += 256) spcw[i] = pcw[i];
    if (tid < RR) { sb3[tid] = b3[tid]; spcb[tid] = pcb[tid]; }
    __syncthreads();

    const int b    = blockIdx.y;
    const int t    = blockIdx.x * 8 + (tid >> 5);   // [0, 4096)
    const int lane = tid & 31;

    // pre tap0 at time t, tap1 (and mus source) at time t+8
    const float* p0 = pre + ((size_t)b * T3L + t) * CC + lane * 8;
    const float* p8 = p0 + 8 * CC;
    float x0[8], x8[8];
    #pragma unroll
    for (int j = 0; j < 8; ++j) { x0[j] = p0[j]; x8[j] = p8[j]; }

    float vfull[RR], mfull[RR];
    #pragma unroll
    for (int r = 0; r < RR; ++r) {
        float vr = 0.f, mr = 0.f;
        const float* wr = sw3  + (r * CC + lane * 8) * 2;
        const float* pr = spcw +  r * CC + lane * 8;
        #pragma unroll
        for (int j = 0; j < 8; ++j) {
            vr += wr[2 * j] * x0[j] + wr[2 * j + 1] * x8[j];
            mr += pr[j] * x8[j];
        }
        #pragma unroll
        for (int off = 16; off > 0; off >>= 1) {
            vr += __shfl_xor(vr, off, 32);
            mr += __shfl_xor(mr, off, 32);
        }
        vfull[r] = vr; mfull[r] = mr;
    }

    if (lane == 0) {
        float hh[RR], ssum = 0.f;
        #pragma unroll
        for (int r = 0; r < RR; ++r) {
            float v = vfull[r] + sb3[r];
            v = (v >= 0.f) ? v : NEG_SLOPE * v;
            hh[r] = v * v + 0.001f;
            ssum += hh[r];
        }
        const float lsum = __logf(ssum);
        const float* g = gumbel + ((size_t)b * TT + t) * RR;
        float a[RR], amax = -1e30f;
        #pragma unroll
        for (int r = 0; r < RR; ++r) {
            a[r] = (__logf(hh[r]) - lsum + g[r]) * (1.f / TAU_F);
            amax = fmaxf(amax, a[r]);
        }
        float z[RR], esum = 0.f;
        #pragma unroll
        for (int r = 0; r < RR; ++r) { z[r] = __expf(a[r] - amax); esum += z[r]; }
        const float einv = 1.f / esum;
        float mus = 0.f;
        #pragma unroll
        for (int r = 0; r < RR; ++r) {
            z[r] *= einv;
            mus  += z[r] * (mfull[r] + spcb[r]);
            out_z[((size_t)b * RR + r) * TT + t] = z[r];
        }
        out_mus[(size_t)b * TT + t] = mus;
    }
}

// ---------------------------------------------------------------------------
// Correction: cov/fish recomputed on the fly from gen_z; x & mus windows in LDS.
// _mus[b,ts] = mus[b,ts+W] - cov(ts+W) * sum_w fish_w(ts+W) * (x[ts+w]-mus[ts+w])
// ---------------------------------------------------------------------------
__global__ void __launch_bounds__(256)
k_corr(const float* __restrict__ x, const float* __restrict__ sct,
       const float* __restrict__ fjt,
       const float* __restrict__ mus, const float* __restrict__ z,
       float* __restrict__ out_umus) {
    __shared__ float sf[RR * WW];
    __shared__ float sx[256 + WW - 1], sm[256 + WW - 1];

    const int tid = threadIdx.x;
    const int b   = blockIdx.y;
    const int ts0 = blockIdx.x * 256;

    for (int i = tid; i < RR * WW; i += 256) sf[i] = fjt[i];
    for (int i = tid; i < 256 + WW - 1; i += 256) {
        int gi = ts0 + i;
        float xv = 0.f, mv = 0.f;
        if (gi < TT) { xv = x[(size_t)b * TT + gi]; mv = mus[(size_t)b * TT + gi]; }
        sx[i] = xv; sm[i] = mv;
    }
    __syncthreads();

    const int ts = ts0 + tid;
    if (ts >= TT - WW) return;
    const int to = ts + WW;

    float zr[RR], cs = 0.f;
    #pragma unroll
    for (int r = 0; r < RR; ++r) {
        zr[r] = z[((size_t)b * RR + r) * TT + to];
        cs   += zr[r] * sct[r];
    }
    const float cov = cs * cs;

    float s = 0.f;
    for (int w = 0; w < WW; ++w) {
        float fw = 0.f;
        #pragma unroll
        for (int r = 0; r < RR; ++r) fw += zr[r] * sf[r * WW + w];
        s += fw * (sx[tid + w] - sm[tid + w]);
    }
    out_umus[(size_t)b * (TT - WW) + ts] = mus[(size_t)b * TT + to] - cov * s;
}

// ---------------------------------------------------------------------------
extern "C" void kernel_launch(void* const* d_in, const int* in_sizes, int n_in,
                              void* d_out, int out_size, void* d_ws, size_t ws_size,
                              hipStream_t stream) {
    const float* x      = (const float*)d_in[0];
    const float* gumbel = (const float*)d_in[1];
    const float* w0     = (const float*)d_in[2];
    const float* w1     = (const float*)d_in[3];
    const float* w2     = (const float*)d_in[4];
    const float* w3     = (const float*)d_in[5];
    const float* b0     = (const float*)d_in[6];
    const float* b1     = (const float*)d_in[7];
    const float* b2     = (const float*)d_in[8];
    const float* b3     = (const float*)d_in[9];
    const float* pcw    = (const float*)d_in[10];
    const float* pcb    = (const float*)d_in[11];
    const float* sct    = (const float*)d_in[12];
    const float* fjt    = (const float*)d_in[13];

    float* out       = (float*)d_out;
    float* out_umus  = out;                        // (16, 3968)
    float* out_mus   = out + BB * (TT - WW);       // (16, 1, 4096)  @ 63488
    float* out_z     = out_mus + BB * TT;          // (16, 8, 4096)  @ 129024

    // Workspace layout (bytes)
    char*   ws    = (char*)d_ws;
    __bf16* wcat1 = (__bf16*)(ws);                                   // 262144
    __bf16* wcat2 = (__bf16*)(ws + 262144);                          // 262144
    __bf16* h1    = (__bf16*)(ws + 524288);                          // 16*4110*256*2
    __bf16* h2    = (__bf16*)(ws + 524288 + 33669120);               // 16*4108*256*2
    float*  pre   = (float*) (ws + 524288 + 33669120 + 33652736);    // 16*4104*256*4

    k_prep_wcat<<<512, 256, 0, stream>>>(w1, wcat1);
    k_prep_wcat<<<512, 256, 0, stream>>>(w2, wcat2);
    k_layer0<<<dim3(T1L, BB), 256, 0, stream>>>(x, w0, b0, h1);
    k_conv_wmma<<<dim3((T2L + 127) / 128, 4, BB), 256, 0, stream>>>(
        h1, wcat1, b1, h2, nullptr, T1L, T2L, 2);
    k_conv_wmma<<<dim3((T3L + 127) / 128, 4, BB), 256, 0, stream>>>(
        h2, wcat2, b2, nullptr, pre, T2L, T3L, 4);
    k_head<<<dim3(TT / 8, BB), 256, 0, stream>>>(
        pre, w3, b3, pcw, pcb, gumbel, out_mus, out_z);
    k_corr<<<dim3((TT - WW + 255) / 256, BB), 256, 0, stream>>>(
        x, sct, fjt, out_mus, out_z, out_umus);
}
